// TimeGNN_33268816675069
// MI455X (gfx1250) — matrix operand order repaired and checked by reading, not compiled, not run on hardware
//
#include <hip/hip_runtime.h>
#include <hip/hip_bf16.h>

typedef __attribute__((ext_vector_type(16))) _Float16 v16h;
typedef __attribute__((ext_vector_type(8)))  float    v8f;
typedef __attribute__((ext_vector_type(4)))  int      v4i;

#define BSZ   64
#define SEQ   128
#define DIM   32
#define HID   64
#define NCLS  18
#define MROWS (BSZ*SEQ)   // 8192

#define TM 128
#define TK 32
#define TN 64

// ---------------------------------------------------------------------------
// Generic GEMM: C[M x 64] (+)= A[M x K] * B[K x 64], f32 in/out, f16 WMMA core.
//  - A(m,k) = A[(m+shiftRows)*lda + k] if 0 <= (m%seqLen)+shiftRows < seqLen else 0
//    (SAME-padded dilated conv taps as row shifts within each batch)
//  - B(k,n) = Bbase[k*sBk + n*sBn], Bbase = B + (m0/rowsPerBatch)*bBatchStride
//  - flags: bit0 = accumulate into C, bit1 = ReLU.  bias (per-n) optional.
// Block = 128 threads (4 waves). Block tile 128x64; each wave owns 32 rows
// (2 A-fragments) x 64 cols (4 B-fragments) -> 8 WMMAs per K-step, B frags
// reused across both A frags.
// ---------------------------------------------------------------------------
__global__ __launch_bounds__(128) void gemm_wmma(
    const float* __restrict__ A, int lda, int shiftRows, int seqLen,
    const float* __restrict__ B, int sBk, int sBn, int bBatchStride, int rowsPerBatch,
    float* __restrict__ C, int ldc, int cColOff,
    const float* __restrict__ bias,
    int K, int flags)
{
    __shared__ _Float16 As[TM * TK];   // [row][k]
    __shared__ _Float16 Bst[TN * TK];  // [n][k]  (transposed so K-pairs are contiguous)

    const int tid  = threadIdx.x;
    const int lane = tid & 31;
    const int wave = tid >> 5;
    const int m0   = blockIdx.x * TM;

    const float* bb = B + ((rowsPerBatch > 0) ? (long)(m0 / rowsPerBatch) * bBatchStride : 0);

    v8f zero = {};
    v8f acc[2][4];
    #pragma unroll
    for (int f = 0; f < 2; ++f)
        #pragma unroll
        for (int i = 0; i < 4; ++i) acc[f][i] = zero;

    // A staging: one full 32-wide row per thread (contiguous 128B -> b128 clause)
    const int aRow = tid;
    const int srow = (m0 + aRow) % seqLen + shiftRows;
    const bool aValid = (unsigned)srow < (unsigned)seqLen;
    const float* aBase = A + (long)(m0 + aRow + shiftRows) * lda;

    for (int k0 = 0; k0 < K; k0 += TK) {
        // ---- stage A tile (128 x 32) ----
        if (aValid) {
            const float* ap = aBase + k0;
            __builtin_prefetch(ap + TK, 0, 1);   // next K-chunk
            #pragma unroll
            for (int q = 0; q < TK; ++q)
                As[aRow * TK + q] = (_Float16)ap[q];
        } else {
            #pragma unroll
            for (int q = 0; q < TK; ++q)
                As[aRow * TK + q] = (_Float16)0.0f;
        }
        // ---- stage B tile (32 x 64) transposed -> Bst[n][k] ----
        if (sBk == 1) {
            // weights stored (N,K): contiguous along k
            const int n  = tid >> 1;
            const int ks = (tid & 1) * 16;
            const float* bp = bb + (long)n * sBn + k0 + ks;
            #pragma unroll
            for (int q = 0; q < 16; ++q)
                Bst[n * TK + ks + q] = (_Float16)bp[q];
        } else if (sBn == 1) {
            // activation B (k-major rows): contiguous along n
            const int kk = tid >> 2;
            const int n0 = (tid & 3) * 16;
            const float* bp = bb + (long)(k0 + kk) * sBk + n0;
            float tmp[16];
            #pragma unroll
            for (int q = 0; q < 16; ++q) tmp[q] = bp[q];
            #pragma unroll
            for (int q = 0; q < 16; ++q)
                Bst[(n0 + q) * TK + kk] = (_Float16)tmp[q];
        } else {
            // generic strided gather: batch loads into regs so they clause
            const int kk = tid >> 2;
            const int n0 = (tid & 3) * 16;
            const float* bp = bb + (long)(k0 + kk) * sBk;
            float tmp[16];
            #pragma unroll
            for (int q = 0; q < 16; ++q) tmp[q] = bp[(long)(n0 + q) * sBn];
            #pragma unroll
            for (int q = 0; q < 16; ++q)
                Bst[(n0 + q) * TK + kk] = (_Float16)tmp[q];
        }
        __syncthreads();

        // ---- build fragments per ISA 16x32 f16 layout ----
        const int r  = lane & 15;
        const int hi = lane >> 4;
        v16h afrag[2];
        #pragma unroll
        for (int f = 0; f < 2; ++f) {
            const unsigned int* arow =
                (const unsigned int*)&As[(wave * 32 + f * 16 + r) * TK];
            #pragma unroll
            for (int v = 0; v < 8; ++v) {
                int kb = ((v & 4) << 2) + hi * 8 + (v & 3) * 2;  // {0|16}+hi*8+2*(v%4)
                ((unsigned int*)&afrag[f])[v] = arow[kb >> 1];
            }
        }
        #pragma unroll
        for (int nt = 0; nt < 4; ++nt) {
            v16h bfrag;
            const unsigned int* brow = (const unsigned int*)&Bst[(nt * 16 + r) * TK];
            #pragma unroll
            for (int v = 0; v < 8; ++v) {
                int kb = ((v & 4) << 2) + hi * 8 + (v & 3) * 2;
                ((unsigned int*)&bfrag)[v] = brow[kb >> 1];
            }
            acc[0][nt] = __builtin_amdgcn_wmma_f32_16x16x32_f16(
                false, afrag[0], false, bfrag, (short)0, acc[0][nt], false, false);
            acc[1][nt] = __builtin_amdgcn_wmma_f32_16x16x32_f16(
                false, afrag[1], false, bfrag, (short)0, acc[1][nt], false, false);
        }
        __syncthreads();
    }

    // ---- epilogue: rows = wave*32 + f*16 + hi*8 + e, col = nt*16 + (lane&15) ----
    const int r  = lane & 15;
    const int hi = lane >> 4;
    #pragma unroll
    for (int f = 0; f < 2; ++f) {
        const int rowBase = m0 + wave * 32 + f * 16 + hi * 8;
        #pragma unroll
        for (int nt = 0; nt < 4; ++nt) {
            int col = nt * 16 + r;
            float bv = bias ? bias[col] : 0.0f;
            #pragma unroll
            for (int e = 0; e < 8; ++e) {
                long off = (long)(rowBase + e) * ldc + cColOff + col;
                float v = acc[f][nt][e] + bv;
                if (flags & 1) v += C[off];
                if (flags & 2) v = fmaxf(v, 0.0f);
                C[off] = v;
            }
        }
    }
}

// ---------------------------------------------------------------------------
// Edge scorer + Gumbel straight-through: writes TRANSPOSED adjacency
// adjT[b][j][i] = 1 if (l0+g0 >= l1+g1) && j > i else 0.
// One block per (b,i), thread j in 0..127.
// ---------------------------------------------------------------------------
__global__ __launch_bounds__(128) void edge_kernel(
    const float* __restrict__ es, const float* __restrict__ er,
    const float* __restrict__ em_b1, const float* __restrict__ em_w2,
    const float* __restrict__ em_b2, const float* __restrict__ gu,
    float* __restrict__ adjT)
{
    __shared__ float ers[HID], w20[HID], w21[HID];
    const int blk = blockIdx.x;
    const int b = blk >> 7;
    const int i = blk & (SEQ - 1);
    const int j = threadIdx.x;
    if (j < HID) {
        ers[j] = er[((long)b * SEQ + i) * HID + j] + em_b1[j];
    } else {
        int h = j - HID;
        w20[h] = em_w2[h];
        w21[h] = em_w2[HID + h];
    }
    __syncthreads();

    const float* esr = es + ((long)b * SEQ + j) * HID;
    float l0 = em_b2[0], l1 = em_b2[1];
    #pragma unroll 8
    for (int h = 0; h < HID; ++h) {
        float v = fmaxf(ers[h] + esr[h], 0.0f);
        l0 += v * w20[h];
        l1 += v * w21[h];
    }
    long gidx = (((long)b * SEQ + i) * SEQ + j) * 2;
    float g0 = -logf(-logf(gu[gidx]));
    float g1 = -logf(-logf(gu[gidx + 1]));
    float a = ((l0 + g0) >= (l1 + g1) && (j > i)) ? 1.0f : 0.0f;
    adjT[((long)b * SEQ + j) * SEQ + i] = a;
}

// Per (b,j) row of adjT: deg = sum_i, fold inv = 1/max(deg,1) into the row.
// Uses CDNA5 async global->LDS copy (raw-byte fit) when available.
__global__ __launch_bounds__(128) void degscale_kernel(float* __restrict__ adjT)
{
    __shared__ float rowbuf[SEQ];
    __shared__ float red[SEQ];
    const long base = (long)blockIdx.x * SEQ;

#if defined(__gfx1250__) && __has_builtin(__builtin_amdgcn_global_load_async_to_lds_b128)
    if (threadIdx.x < 32) {
        // 32 lanes x 16B = 512B = one full 128-float row
        __builtin_amdgcn_global_load_async_to_lds_b128(
            (__attribute__((address_space(1))) v4i*)(adjT + base + threadIdx.x * 4),
            (__attribute__((address_space(3))) v4i*)(rowbuf + threadIdx.x * 4),
            0, 0);
#if __has_builtin(__builtin_amdgcn_s_wait_asynccnt)
        __builtin_amdgcn_s_wait_asynccnt(0);
#else
        asm volatile("s_wait_asynccnt 0x0" ::: "memory");
#endif
    }
    __syncthreads();
    float v = rowbuf[threadIdx.x];
#else
    float v = adjT[base + threadIdx.x];
    (void)rowbuf;
#endif

    red[threadIdx.x] = v;
    __syncthreads();
    for (int s = 64; s > 0; s >>= 1) {
        if (threadIdx.x < s) red[threadIdx.x] += red[threadIdx.x + s];
        __syncthreads();
    }
    float inv = 1.0f / fmaxf(red[0], 1.0f);
    adjT[base + threadIdx.x] = v * inv;
}

__global__ void zero_kernel(float* __restrict__ p, int n)
{
    int i = blockIdx.x * blockDim.x + threadIdx.x;
    if (i < n) p[i] = 0.0f;
}

// Batch-norm stats: sums[0..63]=sum, sums[64..127]=sumsq per channel.
__global__ __launch_bounds__(256) void bnstats_kernel(
    const float* __restrict__ o, float* __restrict__ sums)
{
    __shared__ float ls[HID], ls2[HID];
    if (threadIdx.x < HID) { ls[threadIdx.x] = 0.0f; ls2[threadIdx.x] = 0.0f; }
    __syncthreads();
    const int h = threadIdx.x & (HID - 1);
    const long total = (long)MROWS * HID;
    float s = 0.0f, s2 = 0.0f;
    for (long idx = (long)blockIdx.x * 256 + threadIdx.x; idx < total;
         idx += (long)gridDim.x * 256) {
        float v = o[idx];   // channel == idx & 63 == h (strides are multiples of 64)
        s += v; s2 += v * v;
    }
    atomicAdd(&ls[h], s);
    atomicAdd(&ls2[h], s2);
    __syncthreads();
    if (threadIdx.x < HID) {
        atomicAdd(&sums[threadIdx.x], ls[threadIdx.x]);
        atomicAdd(&sums[HID + threadIdx.x], ls2[threadIdx.x]);
    }
}

// Apply BN, write normalized layer output (in-place over `cur`), stash s=S-1 row.
__global__ __launch_bounds__(256) void bnapply_kernel(
    const float* __restrict__ o, const float* __restrict__ sums,
    const float* __restrict__ bn_g, const float* __restrict__ bn_b,
    float* __restrict__ outbuf, float* __restrict__ lastrow, int k)
{
    long idx = (long)blockIdx.x * 256 + threadIdx.x;
    if (idx >= (long)MROWS * HID) return;
    int h = idx & (HID - 1);
    const float n = (float)MROWS;
    float mu  = sums[h] / n;
    float var = sums[HID + h] / n - mu * mu;
    float v = (o[idx] - mu) * rsqrtf(var + 1e-5f) * bn_g[k * HID + h] + bn_b[k * HID + h];
    outbuf[idx] = v;
    long row = idx >> 6;
    if ((row & (SEQ - 1)) == SEQ - 1) {
        long b = row >> 7;
        lastrow[(long)k * BSZ * HID + b * HID + h] = v;
    }
}

// Final head: gated combine at s=S-1, relu, fce(64->32), relu, out(32->18).
__global__ __launch_bounds__(64) void final_head(
    const float* __restrict__ lastrow, const float* __restrict__ gw_w,
    const float* __restrict__ gw_b, const float* __restrict__ fce_w,
    const float* __restrict__ fce_b, const float* __restrict__ out_w,
    const float* __restrict__ out_b, float* __restrict__ out)
{
    __shared__ float comb[HID], t[HID / 2];
    const int b = blockIdx.x;
    const int tid = threadIdx.x;
    float v = gw_b[0];
    #pragma unroll
    for (int k = 0; k < 3; ++k)
        v += gw_w[k] * lastrow[(long)k * BSZ * HID + b * HID + tid];
    comb[tid] = fmaxf(v, 0.0f);
    __syncthreads();
    if (tid < HID / 2) {
        float a = fce_b[tid];
        #pragma unroll 8
        for (int h = 0; h < HID; ++h) a += comb[h] * fce_w[tid * HID + h];
        t[tid] = fmaxf(a, 0.0f);
    }
    __syncthreads();
    if (tid < NCLS) {
        float a = out_b[tid];
        #pragma unroll
        for (int j = 0; j < HID / 2; ++j) a += t[j] * out_w[tid * (HID / 2) + j];
        out[(long)b * NCLS + tid] = a;
    }
}

// ---------------------------------------------------------------------------
extern "C" void kernel_launch(void* const* d_in, const int* in_sizes, int n_in,
                              void* d_out, int out_size, void* d_ws, size_t ws_size,
                              hipStream_t stream)
{
    const float* x        = (const float*)d_in[0];
    const float* gu       = (const float*)d_in[1];
    const float* c11_w    = (const float*)d_in[2];
    const float* c11_b    = (const float*)d_in[3];
    const float* c12_w    = (const float*)d_in[4];
    const float* c12_b    = (const float*)d_in[5];
    const float* c21_w    = (const float*)d_in[6];
    const float* c21_b    = (const float*)d_in[7];
    const float* c22_w    = (const float*)d_in[8];
    const float* c22_b    = (const float*)d_in[9];
    const float* c31_w    = (const float*)d_in[10];
    const float* c31_b    = (const float*)d_in[11];
    const float* fcf_w    = (const float*)d_in[12];
    const float* fcf_b    = (const float*)d_in[13];
    const float* em_w1    = (const float*)d_in[14];
    const float* em_b1    = (const float*)d_in[15];
    const float* em_w2    = (const float*)d_in[16];
    const float* em_b2    = (const float*)d_in[17];
    const float* sage_l_w = (const float*)d_in[18];
    const float* sage_l_b = (const float*)d_in[19];
    const float* sage_r_w = (const float*)d_in[20];
    const float* bn_g     = (const float*)d_in[21];
    const float* bn_b     = (const float*)d_in[22];
    const float* gw_w     = (const float*)d_in[23];
    const float* gw_b     = (const float*)d_in[24];
    const float* fce_w    = (const float*)d_in[25];
    const float* fce_b    = (const float*)d_in[26];
    const float* out_w    = (const float*)d_in[27];
    const float* out_b    = (const float*)d_in[28];
    float* out = (float*)d_out;

    // workspace layout (floats); total ~6.3M floats ~ 25 MB
    float* ws   = (float*)d_ws;
    float* cat  = ws;                          // 8192*192
    float* p1   = cat + (long)MROWS * 192;     // 8192*64
    float* p2   = p1 + (long)MROWS * HID;
    float* hbuf = p2 + (long)MROWS * HID;      // h / out_cur (B,S,H)
    float* es   = hbuf + (long)MROWS * HID;
    float* er   = es + (long)MROWS * HID;
    float* adjT = er + (long)MROWS * HID;      // (B,S,S) = 1,048,576
    float* aggr = adjT + (long)BSZ * SEQ * SEQ;
    float* opre = aggr + (long)MROWS * HID;
    float* last = opre + (long)MROWS * HID;    // 3*64*64
    float* sums = last + 3L * BSZ * HID;       // 128

    const dim3 gG(MROWS / TM), bG(128);

    // ---- pointwise convs (D=32 -> H=64) ----
    gemm_wmma<<<gG, bG, 0, stream>>>(x, DIM, 0, SEQ, c11_w, 1, DIM, 0, 0,
                                     p1, HID, 0, c11_b, DIM, 0);
    gemm_wmma<<<gG, bG, 0, stream>>>(x, DIM, 0, SEQ, c21_w, 1, DIM, 0, 0,
                                     p2, HID, 0, c21_b, DIM, 0);
    gemm_wmma<<<gG, bG, 0, stream>>>(x, DIM, 0, SEQ, c31_w, 1, DIM, 0, 0,
                                     cat, 192, 128, c31_b, DIM, 0);   // f3 -> cols 128..191

    // ---- dilated conv f1: taps at (t-1)*3 on p1, weight (H,H,3) -> cols 0..63 ----
    for (int t = 0; t < 3; ++t)
        gemm_wmma<<<gG, bG, 0, stream>>>(p1, HID, (t - 1) * 3, SEQ,
                                         c12_w + t, 3, 192, 0, 0,
                                         cat, 192, 0, t == 0 ? c12_b : nullptr,
                                         HID, t == 0 ? 0 : 1);
    // ---- dilated conv f2: taps at (t-2)*5 on p2, weight (H,H,5) -> cols 64..127 ----
    for (int t = 0; t < 5; ++t)
        gemm_wmma<<<gG, bG, 0, stream>>>(p2, HID, (t - 2) * 5, SEQ,
                                         c22_w + t, 5, 320, 0, 0,
                                         cat, 192, 64, t == 0 ? c22_b : nullptr,
                                         HID, t == 0 ? 0 : 1);

    // ---- h = relu(cat @ fcf_w.T + fcf_b) ----
    gemm_wmma<<<gG, bG, 0, stream>>>(cat, 192, 0, SEQ, fcf_w, 1, 192, 0, 0,
                                     hbuf, HID, 0, fcf_b, 192, 2);

    // ---- e_s = h @ Ws.T, e_r = h @ Wr.T (halves of em_w1 (H,2H)) ----
    gemm_wmma<<<gG, bG, 0, stream>>>(hbuf, HID, 0, SEQ, em_w1, 1, 2 * HID, 0, 0,
                                     es, HID, 0, nullptr, HID, 0);
    gemm_wmma<<<gG, bG, 0, stream>>>(hbuf, HID, 0, SEQ, em_w1 + HID, 1, 2 * HID, 0, 0,
                                     er, HID, 0, nullptr, HID, 0);

    // ---- edges + gumbel argmax -> adjT; fold 1/deg into adjT ----
    edge_kernel<<<BSZ * SEQ, 128, 0, stream>>>(es, er, em_b1, em_w2, em_b2, gu, adjT);
    degscale_kernel<<<BSZ * SEQ, 128, 0, stream>>>(adjT);

    // ---- GNN layers ----
    float* cur = hbuf;
    for (int k = 0; k < 3; ++k) {
        // aggr[b,j,:] = sum_i adjT[b,j,i] * cur[b,i,:]  (per-batch GEMM, batched B)
        gemm_wmma<<<gG, bG, 0, stream>>>(adjT, SEQ, 0, SEQ,
                                         cur, HID, 1, SEQ * HID, SEQ,
                                         aggr, HID, 0, nullptr, SEQ, 0);
        // o = aggr @ Wl.T + bl
        gemm_wmma<<<gG, bG, 0, stream>>>(aggr, HID, 0, SEQ,
                                         sage_l_w + (long)k * HID * HID, 1, HID, 0, 0,
                                         opre, HID, 0, sage_l_b + (long)k * HID,
                                         HID, 0);
        // o += cur @ Wr.T
        gemm_wmma<<<gG, bG, 0, stream>>>(cur, HID, 0, SEQ,
                                         sage_r_w + (long)k * HID * HID, 1, HID, 0, 0,
                                         opre, HID, 0, nullptr, HID, 1);
        // batch norm over all B*S rows
        zero_kernel<<<1, 128, 0, stream>>>(sums, 128);
        bnstats_kernel<<<64, 256, 0, stream>>>(opre, sums);
        bnapply_kernel<<<(MROWS * HID) / 256, 256, 0, stream>>>(
            opre, sums, bn_g, bn_b, cur, last, k);
    }

    // ---- head ----
    final_head<<<BSZ, 64, 0, stream>>>(last, gw_w, gw_b, fce_w, fce_b,
                                       out_w, out_b, out);
}
